// DynamicClustering_26938034880969
// MI455X (gfx1250) — matrix operand order repaired and checked by reading, not compiled, not run on hardware
//
#include <hip/hip_runtime.h>
#include <hip/hip_bf16.h>

typedef __attribute__((ext_vector_type(16))) _Float16 v16h;
typedef __attribute__((ext_vector_type(8)))  _Float16 v8h;
typedef __attribute__((ext_vector_type(8)))  float    v8f;

#define BB 4
#define NN 4096
#define CC 768
#define CN 41          // ceil(4096*0.01)
#define FINF 1e30f

// ---------------- zero init (d_out poisoned 0xAA; atomics need zeros) ----------------
__global__ void dc_zero(float* out, int n_out, float* aw, unsigned* dmax) {
    int i = blockIdx.x * 256 + threadIdx.x;
    if (i < n_out) out[i] = 0.0f;
    if (i < BB * CN) aw[i] = 0.0f;
    if (i < BB) dmax[i] = 0u;
}

// ---------------- prep: f16 copy, row |x|^2, token_weight = exp(x.W + b) ----------------
__global__ __launch_bounds__(256) void dc_prep(const float* __restrict__ x,
                                               const float* __restrict__ W,
                                               const float* __restrict__ bb,
                                               _Float16* __restrict__ xh,
                                               float* __restrict__ sq,
                                               float* __restrict__ tw) {
    int bn = blockIdx.x;
    const float* xn = x + (size_t)bn * CC;
    _Float16* xhn = xh + (size_t)bn * CC;
    int t = threadIdx.x;
    float psq = 0.0f, pdot = 0.0f;
    for (int c = t; c < CC; c += 256) {
        float v = xn[c];
        xhn[c] = (_Float16)v;
        psq  = fmaf(v, v, psq);
        pdot = fmaf(v, W[c], pdot);
    }
    __shared__ float s1[256], s2[256];
    s1[t] = psq; s2[t] = pdot;
    __syncthreads();
    for (int s = 128; s > 0; s >>= 1) {
        if (t < s) { s1[t] += s1[t + s]; s2[t] += s2[t + s]; }
        __syncthreads();
    }
    if (t == 0) { sq[bn] = s1[0]; tw[bn] = expf(s2[0] + bb[0]); }
}

// ---------------- WMMA strip helper: 16x64 dist strip (4 independent accumulators) ----------------
__device__ __forceinline__ void dc_dist_strip(const _Float16* __restrict__ xb,
                                              const float* __restrict__ sqb,
                                              const float* __restrict__ s_sqi,
                                              int i0, int j0, int lane,
                                              float (&s_tile)[16][65], float& tmax) {
    const int nloc  = lane & 15;
    const int half  = lane >> 4;           // 0 or 1
    const int mbase = half * 8;
    const int aoff  = half * 8;            // A: K sub-offset per half-wave
    const _Float16* arow  = xb + (size_t)(i0 + nloc) * CC;
    const _Float16* brow0 = xb + (size_t)(j0 + nloc) * CC + half * 8;

    v8f acc[4] = {{}, {}, {}, {}};
    for (int kk = 0; kk < CC; kk += 32) {
        v8h lo = *(const v8h*)(arow + kk + aoff);
        v8h hi = *(const v8h*)(arow + kk + 16 + aoff);
        v16h A;
#pragma unroll
        for (int e = 0; e < 8; e++) { A[e] = lo[e]; A[e + 8] = hi[e]; }
#pragma unroll
        for (int t = 0; t < 4; t++) {
            v8h blo = *(const v8h*)(brow0 + (size_t)t * 16 * CC + kk);
            v8h bhi = *(const v8h*)(brow0 + (size_t)t * 16 * CC + kk + 16);
            v16h Bv;
#pragma unroll
            for (int e = 0; e < 8; e++) { Bv[e] = blo[e]; Bv[e + 8] = bhi[e]; }
            acc[t] = __builtin_amdgcn_wmma_f32_16x16x32_f16(false, A, false, Bv,
                                                            (short)0, acc[t], false, false);
        }
    }
    const float invC = 1.0f / (float)CC;
#pragma unroll
    for (int t = 0; t < 4; t++) {
        float sqj = sqb[j0 + 16 * t + nloc];
#pragma unroll
        for (int v = 0; v < 8; v++) {
            float d2 = (s_sqi[mbase + v] + sqj - 2.0f * acc[t][v]) * invC;
            float d = sqrtf(fmaxf(d2, 0.0f));
            tmax = fmaxf(tmax, d);
            s_tile[mbase + v][16 * t + nloc] = d;   // D layout: VGPR v -> M = mbase+v
        }
    }
}

// ---------------- pass 1: kNN(5) density + per-batch dist_max ----------------
__global__ __launch_bounds__(32) void dc_pass1(const _Float16* __restrict__ xh,
                                               const float* __restrict__ sq,
                                               float* __restrict__ density,
                                               unsigned* __restrict__ dmax) {
    int b = blockIdx.y;
    int i0 = blockIdx.x * 16;
    int lane = threadIdx.x;
    const _Float16* xb = xh + (size_t)b * NN * CC;
    const float* sqb = sq + (size_t)b * NN;

    __shared__ float s_sqi[16];
    __shared__ float s_tile[16][65];
    if (lane < 16) s_sqi[lane] = sqb[i0 + lane];
    __syncthreads();

    float tk[5] = {FINF, FINF, FINF, FINF, FINF};
    float tmax = 0.0f;
    for (int j0 = 0; j0 < NN; j0 += 64) {
        dc_dist_strip(xb, sqb, s_sqi, i0, j0, lane, s_tile, tmax);
        __syncthreads();
        if (lane < 16) {
            for (int c2 = 0; c2 < 64; c2++) {
                float d = s_tile[lane][c2];
                if (d < tk[4]) {
                    tk[4] = d;
#pragma unroll
                    for (int j = 4; j > 0; j--) {
                        if (tk[j] < tk[j - 1]) { float t = tk[j]; tk[j] = tk[j - 1]; tk[j - 1] = t; }
                    }
                }
            }
        }
        __syncthreads();
    }
    for (int s = 16; s > 0; s >>= 1) tmax = fmaxf(tmax, __shfl_xor(tmax, s, 32));
    if (lane == 0) atomicMax(&dmax[b], __float_as_uint(tmax));  // dist >= 0: bit-order == value-order
    if (lane < 16) {
        float m2 = (tk[0]*tk[0] + tk[1]*tk[1] + tk[2]*tk[2] + tk[3]*tk[3] + tk[4]*tk[4]) * 0.2f;
        density[(size_t)b * NN + i0 + lane] = expf(-m2);
    }
}

// ---------------- pass 2: min dist to any strictly-denser token; score ----------------
__global__ __launch_bounds__(32) void dc_pass2(const _Float16* __restrict__ xh,
                                               const float* __restrict__ sq,
                                               const float* __restrict__ density,
                                               const unsigned* __restrict__ dmax,
                                               float* __restrict__ score) {
    int b = blockIdx.y;
    int i0 = blockIdx.x * 16;
    int lane = threadIdx.x;
    const _Float16* xb = xh + (size_t)b * NN * CC;
    const float* sqb = sq + (size_t)b * NN;
    const float* rho = density + (size_t)b * NN;

    __shared__ float s_sqi[16];
    __shared__ float s_tile[16][65];
    __shared__ float s_dcol[64];
    if (lane < 16) s_sqi[lane] = sqb[i0 + lane];
    __syncthreads();

    float myrho = (lane < 16) ? rho[i0 + lane] : 0.0f;
    float dmin = FINF;
    float tmax = 0.0f;  // unused here
    for (int j0 = 0; j0 < NN; j0 += 64) {
        s_dcol[lane]      = rho[j0 + lane];
        s_dcol[lane + 32] = rho[j0 + lane + 32];
        dc_dist_strip(xb, sqb, s_sqi, i0, j0, lane, s_tile, tmax);
        __syncthreads();
        if (lane < 16) {
            for (int c2 = 0; c2 < 64; c2++) {
                if (s_dcol[c2] > myrho) dmin = fminf(dmin, s_tile[lane][c2]);
            }
        }
        __syncthreads();
    }
    if (lane < 16) {
        dmin = fminf(dmin, __uint_as_float(dmax[b]));
        score[(size_t)b * NN + i0 + lane] = dmin * myrho;
    }
}

// ---------------- top-41 per batch (iterative argmax; mutates score, rebuilt each call) ----------------
__global__ __launch_bounds__(256) void dc_topk(float* __restrict__ score, int* __restrict__ idxd) {
    int b = blockIdx.x, t = threadIdx.x;
    __shared__ float sv[256];
    __shared__ int   si[256];
    float* sc = score + (size_t)b * NN;
    for (int it = 0; it < CN; it++) {
        float bv = -FINF; int bi = NN;
        for (int i = t; i < NN; i += 256) {
            float v = sc[i];
            if (v > bv || (v == bv && i < bi)) { bv = v; bi = i; }
        }
        sv[t] = bv; si[t] = bi;
        __syncthreads();
        for (int s = 128; s > 0; s >>= 1) {
            if (t < s) {
                if (sv[t + s] > sv[t] || (sv[t + s] == sv[t] && si[t + s] < si[t])) {
                    sv[t] = sv[t + s]; si[t] = si[t + s];
                }
            }
            __syncthreads();
        }
        if (t == 0) { idxd[b * CN + it] = si[0]; sc[si[0]] = -FINF; }
        __syncthreads();
    }
}

// ---------------- assign each token to nearest of 41 centers ----------------
__global__ __launch_bounds__(64) void dc_assign(const float* __restrict__ x,
                                                const float* __restrict__ sq,
                                                const int* __restrict__ idxd,
                                                int* __restrict__ idxc) {
    int b = blockIdx.y, n = blockIdx.x, t = threadIdx.x;
    __shared__ float sd[64];
    __shared__ int   sc[64];
    const float* xn = x + ((size_t)b * NN + n) * CC;
    float d2 = FINF;
    if (t < CN) {
        int ci = idxd[b * CN + t];
        const float* xc = x + ((size_t)b * NN + ci) * CC;
        float dot = 0.0f;
        for (int c = 0; c < CC; c++) dot = fmaf(xn[c], xc[c], dot);
        d2 = sq[(size_t)b * NN + n] + sq[(size_t)b * NN + ci] - 2.0f * dot;
    }
    sd[t] = d2; sc[t] = t;
    __syncthreads();
    for (int s = 32; s > 0; s >>= 1) {
        if (t < s) {
            if (sd[t + s] < sd[t] || (sd[t + s] == sd[t] && sc[t + s] < sc[t])) {
                sd[t] = sd[t + s]; sc[t] = sc[t + s];
            }
        }
        __syncthreads();
    }
    if (t == 0) idxc[(size_t)b * NN + n] = sc[0];
}

__global__ void dc_fix_centers(const int* __restrict__ idxd, int* __restrict__ idxc) {
    int j = blockIdx.x * 64 + threadIdx.x;
    if (j < BB * CN) { int b = j / CN, c = j % CN; idxc[(size_t)b * NN + idxd[j]] = c; }
}

// ---------------- weighted merge ----------------
__global__ void dc_wsum(const float* __restrict__ tw, const int* __restrict__ idxc,
                        float* __restrict__ aw) {
    int i = blockIdx.x * 256 + threadIdx.x;
    if (i < BB * NN) { int b = i >> 12; atomicAdd(&aw[b * CN + idxc[i]], tw[i]); }
}

__global__ __launch_bounds__(256) void dc_merge(const float* __restrict__ x,
                                                const float* __restrict__ tw,
                                                const int* __restrict__ idxc,
                                                const float* __restrict__ aw,
                                                float* __restrict__ out) {
    int bn = blockIdx.x;
    int b = bn >> 12;                  // N = 4096
    int cid = idxc[bn];
    float w = tw[bn] / (aw[b * CN + cid] + 1e-6f);
    const float* xn = x + (size_t)bn * CC;
    float* o = out + ((size_t)b * CN + cid) * CC;
    for (int c = threadIdx.x; c < CC; c += 256) atomicAdd(&o[c], xn[c] * w);
}

extern "C" void kernel_launch(void* const* d_in, const int* in_sizes, int n_in,
                              void* d_out, int out_size, void* d_ws, size_t ws_size,
                              hipStream_t stream) {
    const float* x  = (const float*)d_in[0];   // [B,N,C] fp32
    // d_in[1] anomaly_map: unused by the reference
    const float* W  = (const float*)d_in[2];   // [1,C]
    const float* bv = (const float*)d_in[3];   // [1]
    float* out = (float*)d_out;                // [B,CN,C]

    // workspace layout (256B aligned chunks)
    char* ws = (char*)d_ws;
    size_t off = 0;
    auto take = [&](size_t bytes) { char* p = ws + off; off = (off + bytes + 255) & ~(size_t)255; return p; };
    _Float16* xh   = (_Float16*)take((size_t)BB * NN * CC * sizeof(_Float16)); // 24 MB
    float*    sq   = (float*)take((size_t)BB * NN * sizeof(float));
    float*    tw   = (float*)take((size_t)BB * NN * sizeof(float));
    float*    rho  = (float*)take((size_t)BB * NN * sizeof(float));
    float*    scr  = (float*)take((size_t)BB * NN * sizeof(float));
    unsigned* dmx  = (unsigned*)take(BB * sizeof(unsigned));
    int*      idxd = (int*)take(BB * CN * sizeof(int));
    int*      idxc = (int*)take((size_t)BB * NN * sizeof(int));
    float*    aw   = (float*)take(BB * CN * sizeof(float));

    const int n_out = BB * CN * CC;
    dc_zero<<<(n_out + 255) / 256, 256, 0, stream>>>(out, n_out, aw, dmx);
    dc_prep<<<BB * NN, 256, 0, stream>>>(x, W, bv, xh, sq, tw);
    dc_pass1<<<dim3(NN / 16, BB), 32, 0, stream>>>(xh, sq, rho, dmx);
    dc_pass2<<<dim3(NN / 16, BB), 32, 0, stream>>>(xh, sq, rho, dmx, scr);
    dc_topk<<<BB, 256, 0, stream>>>(scr, idxd);
    dc_assign<<<dim3(NN, BB), 64, 0, stream>>>(x, sq, idxd, idxc);
    dc_fix_centers<<<(BB * CN + 63) / 64, 64, 0, stream>>>(idxd, idxc);
    dc_wsum<<<(BB * NN + 255) / 256, 256, 0, stream>>>(tw, idxc, aw);
    dc_merge<<<BB * NN, 256, 0, stream>>>(x, tw, idxc, aw, out);
    (void)in_sizes; (void)n_in; (void)out_size; (void)ws_size;
}
